// LAHRv3_40192303956232
// MI455X (gfx1250) — compile-verified, matrix operand order checked by reading, not run on hardware
//
#include <hip/hip_runtime.h>
#include <hip/hip_bf16.h>
#include <math.h>

// ---------------------------------------------------------------------------
// Model dims (match reference)
// ---------------------------------------------------------------------------
#define BB   4
#define TT   512
#define DD   768
#define HH   12
#define LL   12
#define HD   64
#define DFF  2048
#define VOC  50257
#define NMEM 1024
#define TOPK 8
#define NLAT 4
#define CAP  64

typedef __attribute__((ext_vector_type(16))) __bf16 v16bf;
typedef __attribute__((ext_vector_type(8)))  float  v8f;

// ---------------------------------------------------------------------------
// WMMA bf16 GEMM:  C[M,N] = alpha * A[M,K] x B    (B is [N,K] if BT, else [K,N])
// Batched over z = zb*Hdim + zh with independent strides for zb / zh so we can
// index straight into the interleaved [B,T,3,H,HD] qkv buffer.
//
// Register-blocked: one wave computes a 32(M) x 64(N) C region = 2x4 WMMA
// tiles, K stepped by 32.  Per K-step: 2 A-tile loads + 4 B-tile loads feed
// 8 independent v_wmma_f32_16x16x32_bf16 issues (A reused 4x, B reused 2x).
// A uniform fast path (full tile in bounds -- all tiles except the LM-head
// N edge) keeps the hot loop free of per-lane EXEC masking and zero-fills.
//
// A-operand layout (16-bit A 16x32): lane m = lane%16, half = lane/16,
//   VGPR j in 0..3 -> K = k0 + 2j + 8*half ; VGPR j in 4..7 -> +16.
// B-operand uses the symmetric layout with lane = N column.
// C/D layout: lane n = lane%16, VGPR r -> M = r + 8*(lane/16).
// ---------------------------------------------------------------------------
__device__ __forceinline__ v16bf load_k32_rowmajor(const float* __restrict__ row,
                                                   int k0, int half) {
  v16bf v;
#pragma unroll
  for (int j = 0; j < 4; ++j) {
    const float2 f = *(const float2*)(row + k0 + 2 * j + 8 * half);
    const float2 g = *(const float2*)(row + k0 + 16 + 2 * j + 8 * half);
    v[2 * j]     = (__bf16)f.x;  v[2 * j + 1]     = (__bf16)f.y;
    v[8 + 2 * j] = (__bf16)g.x;  v[8 + 2 * j + 1] = (__bf16)g.y;
  }
  return v;
}

__device__ __forceinline__ v16bf load_k32_colmajor(const float* __restrict__ col,
                                                   int k0, int half, int ldb) {
  v16bf v;
#pragma unroll
  for (int j = 0; j < 4; ++j) {
    const int k1 = k0 + 2 * j + 8 * half;
    const int k2 = k0 + 16 + 2 * j + 8 * half;
    v[2 * j]         = (__bf16)col[(size_t)k1 * ldb];
    v[2 * j + 1]     = (__bf16)col[(size_t)(k1 + 1) * ldb];
    v[8 + 2 * j]     = (__bf16)col[(size_t)k2 * ldb];
    v[8 + 2 * j + 1] = (__bf16)col[(size_t)(k2 + 1) * ldb];
  }
  return v;
}

__device__ __forceinline__ v16bf zero16bf() {
  v16bf z;
#pragma unroll
  for (int i = 0; i < 16; ++i) z[i] = (__bf16)0.0f;
  return z;
}

template <bool BT>
__global__ __launch_bounds__(32) void wmma_gemm_kernel(
    const float* __restrict__ A, const float* __restrict__ B,
    float* __restrict__ C, int M, int N, int K, int lda, int ldb, int ldc,
    long long sAb, long long sAh, long long sBb, long long sBh,
    long long sCb, long long sCh, int Hdim, float alpha) {
  const int lane = threadIdx.x & 31;
  const int half = lane >> 4;
  const int lr   = lane & 15;
  const int tn = blockIdx.x;   // 64-wide N macro-tile
  const int tm = blockIdx.y;   // 32-wide M macro-tile
  const int z  = blockIdx.z;
  const int zb = z / Hdim, zh = z % Hdim;

  const float* Ab = A + (size_t)zb * sAb + (size_t)zh * sAh;
  const float* Bb = B + (size_t)zb * sBb + (size_t)zh * sBh;
  float*       Cb = C + (size_t)zb * sCb + (size_t)zh * sCh;

  const int gm0 = tm * 32 + lr;
  const int gm1 = gm0 + 16;
  const float* Arow0 = Ab + (size_t)gm0 * lda;
  const float* Arow1 = Ab + (size_t)gm1 * lda;

  int gnB[4];
  const float* Brow[4];
#pragma unroll
  for (int t = 0; t < 4; ++t) {
    gnB[t]  = tn * 64 + t * 16 + lr;
    Brow[t] = Bb + (size_t)gnB[t] * (BT ? (size_t)ldb : (size_t)1);
  }

  v8f acc[2][4] = {};

  // Uniform (wave-invariant) full-tile test: no per-lane masking needed.
  const bool full = (tm * 32 + 32 <= M) && (tn * 64 + 64 <= N);

  if (full) {
    for (int k0 = 0; k0 < K; k0 += 32) {
      // Speculative prefetch of next K chunk (OOB prefetch is silently
      // dropped in translation -- ISA 10.5), so no end-of-loop guard.
      __builtin_prefetch(Arow0 + k0 + 32, 0, 1);
      __builtin_prefetch(Arow1 + k0 + 32, 0, 1);

      const v16bf a0 = load_k32_rowmajor(Arow0, k0, half);
      const v16bf a1 = load_k32_rowmajor(Arow1, k0, half);
      v16bf b[4];
#pragma unroll
      for (int t = 0; t < 4; ++t)
        b[t] = BT ? load_k32_rowmajor(Brow[t], k0, half)
                  : load_k32_colmajor(Brow[t], k0, half, ldb);

#pragma unroll
      for (int t = 0; t < 4; ++t) {
        acc[0][t] = __builtin_amdgcn_wmma_f32_16x16x32_bf16(
            false, a0, false, b[t], (short)0, acc[0][t], false, false);
        acc[1][t] = __builtin_amdgcn_wmma_f32_16x16x32_bf16(
            false, a1, false, b[t], (short)0, acc[1][t], false, false);
      }
    }
#pragma unroll
    for (int i = 0; i < 2; ++i) {
#pragma unroll
      for (int t = 0; t < 4; ++t) {
        const int gn = tn * 64 + t * 16 + lr;
#pragma unroll
        for (int r = 0; r < 8; ++r) {
          const int gm = tm * 32 + i * 16 + r + 8 * half;
          Cb[(size_t)gm * ldc + gn] = alpha * acc[i][t][r];
        }
      }
    }
  } else {
    // Edge path (LM-head N boundary): per-lane guards, zero padding.
    const bool am0 = (gm0 < M);
    const bool am1 = (gm1 < M);
    bool bn[4];
#pragma unroll
    for (int t = 0; t < 4; ++t) bn[t] = (gnB[t] < N);

    for (int k0 = 0; k0 < K; k0 += 32) {
      const v16bf a0 = am0 ? load_k32_rowmajor(Arow0, k0, half) : zero16bf();
      const v16bf a1 = am1 ? load_k32_rowmajor(Arow1, k0, half) : zero16bf();
      v16bf b[4];
#pragma unroll
      for (int t = 0; t < 4; ++t) {
        if (bn[t])
          b[t] = BT ? load_k32_rowmajor(Brow[t], k0, half)
                    : load_k32_colmajor(Brow[t], k0, half, ldb);
        else
          b[t] = zero16bf();
      }
#pragma unroll
      for (int t = 0; t < 4; ++t) {
        acc[0][t] = __builtin_amdgcn_wmma_f32_16x16x32_bf16(
            false, a0, false, b[t], (short)0, acc[0][t], false, false);
        acc[1][t] = __builtin_amdgcn_wmma_f32_16x16x32_bf16(
            false, a1, false, b[t], (short)0, acc[1][t], false, false);
      }
    }
#pragma unroll
    for (int i = 0; i < 2; ++i) {
#pragma unroll
      for (int t = 0; t < 4; ++t) {
        const int gn = tn * 64 + t * 16 + lr;
#pragma unroll
        for (int r = 0; r < 8; ++r) {
          const int gm = tm * 32 + i * 16 + r + 8 * half;
          if (gm < M && gn < N) Cb[(size_t)gm * ldc + gn] = alpha * acc[i][t][r];
        }
      }
    }
  }
}

static void gemm_nt(hipStream_t s, const float* A, const float* B, float* C,
                    int M, int N, int K, int lda, int ldb, int ldc,
                    long long sAb, long long sAh, long long sBb, long long sBh,
                    long long sCb, long long sCh, int Z, int Hdim, float alpha) {
  dim3 g((N + 63) / 64, (M + 31) / 32, Z);
  wmma_gemm_kernel<true><<<g, 32, 0, s>>>(A, B, C, M, N, K, lda, ldb, ldc,
                                          sAb, sAh, sBb, sBh, sCb, sCh, Hdim, alpha);
}
static void gemm_nn(hipStream_t s, const float* A, const float* B, float* C,
                    int M, int N, int K, int lda, int ldb, int ldc,
                    long long sAb, long long sAh, long long sBb, long long sBh,
                    long long sCb, long long sCh, int Z, int Hdim, float alpha) {
  dim3 g((N + 63) / 64, (M + 31) / 32, Z);
  wmma_gemm_kernel<false><<<g, 32, 0, s>>>(A, B, C, M, N, K, lda, ldb, ldc,
                                           sAb, sAh, sBb, sBh, sCb, sCh, Hdim, alpha);
}

// ---------------------------------------------------------------------------
// Elementwise / reduction kernels
// ---------------------------------------------------------------------------
__global__ void embed_kernel(const int* __restrict__ ids,
                             const float* __restrict__ emb,
                             const float* __restrict__ pos,
                             float* __restrict__ x) {
  const int bt = blockIdx.x;
  const int t  = bt % TT;
  const int id = ids[bt];
  const float* e = emb + (size_t)id * DD;
  const float* p = pos + (size_t)t * DD;
  float* o = x + (size_t)bt * DD;
  for (int i = threadIdx.x; i < DD; i += blockDim.x) o[i] = e[i] + p[i];
}

__global__ void rmsnorm_kernel(const float* __restrict__ x,
                               const float* __restrict__ w,
                               float* __restrict__ y, int D) {
  const int row = blockIdx.x;
  const int tid = threadIdx.x;
  const float* xr = x + (size_t)row * D;
  float s = 0.f;
  for (int i = tid; i < D; i += blockDim.x) { const float v = xr[i]; s += v * v; }
  __shared__ float red[256];
  red[tid] = s; __syncthreads();
  for (int o = 128; o > 0; o >>= 1) {
    if (tid < o) red[tid] += red[tid + o];
    __syncthreads();
  }
  const float inv = rsqrtf(red[0] / (float)D + 1e-6f);
  float* yr = y + (size_t)row * D;
  for (int i = tid; i < D; i += blockDim.x) yr[i] = xr[i] * inv * w[i];
}

__global__ void softmax_causal_kernel(float* __restrict__ sc) {
  const int row = blockIdx.x;          // (b*H + h)*T + q
  const int q   = row % TT;
  const int tid = threadIdx.x;
  float* r = sc + (size_t)row * TT;
  __shared__ float red[256];
  float mx = -3.0e38f;
  for (int k = tid; k <= q; k += blockDim.x) mx = fmaxf(mx, r[k]);
  red[tid] = mx; __syncthreads();
  for (int o = 128; o > 0; o >>= 1) {
    if (tid < o) red[tid] = fmaxf(red[tid], red[tid + o]);
    __syncthreads();
  }
  mx = red[0]; __syncthreads();
  float sum = 0.f;
  for (int k = tid; k < TT; k += blockDim.x) {
    const float e = (k <= q) ? __expf(r[k] - mx) : 0.f;
    r[k] = e; sum += e;
  }
  red[tid] = sum; __syncthreads();
  for (int o = 128; o > 0; o >>= 1) {
    if (tid < o) red[tid] += red[tid + o];
    __syncthreads();
  }
  const float inv = 1.f / red[0];
  for (int k = tid; k < TT; k += blockDim.x) r[k] *= inv;
}

__global__ void add_inplace_kernel(float* __restrict__ dst,
                                   const float* __restrict__ src, long long n) {
  const long long i = (long long)blockIdx.x * blockDim.x + threadIdx.x;
  if (i < n) dst[i] += src[i];
}

__global__ void silu_mul_kernel(float* __restrict__ h1,
                                const float* __restrict__ h2, long long n) {
  const long long i = (long long)blockIdx.x * blockDim.x + threadIdx.x;
  if (i < n) {
    const float a = h1[i];
    h1[i] = (a / (1.f + __expf(-a))) * h2[i];
  }
}

__global__ void rowdot_kernel(const float* __restrict__ x,
                              const float* __restrict__ w,
                              float* __restrict__ out, int D) {
  const int row = blockIdx.x;
  const int tid = threadIdx.x;
  const float* xr = x + (size_t)row * D;
  float s = 0.f;
  for (int i = tid; i < D; i += blockDim.x) s += xr[i] * w[i];
  __shared__ float red[128];
  red[tid] = s; __syncthreads();
  for (int o = 64; o > 0; o >>= 1) {
    if (tid < o) red[tid] += red[tid + o];
    __syncthreads();
  }
  if (tid == 0) out[row] = red[0];
}

// Exact top-k membership via rank counting (lowest-index tie break = jax top_k).
__global__ void mod_select_kernel(const float* __restrict__ sc,
                                  int* __restrict__ sel) {
  const int b = blockIdx.x / TT;
  const int t = blockIdx.x % TT;
  const int tid = threadIdx.x;
  const float* s = sc + (size_t)b * TT;
  const float st = s[t];
  int cnt = 0;
  for (int k = tid; k < TT; k += blockDim.x) {
    const float v = s[k];
    if (v > st || (v == st && k < t)) ++cnt;
  }
  __shared__ int red[128];
  red[tid] = cnt; __syncthreads();
  for (int o = 64; o > 0; o >>= 1) {
    if (tid < o) red[tid] += red[tid + o];
    __syncthreads();
  }
  if (tid == 0) sel[blockIdx.x] = (red[0] < CAP) ? 1 : 0;
}

__global__ void mod_merge_kernel(float* __restrict__ x,
                                 const float* __restrict__ y,
                                 const int* __restrict__ sel) {
  const int row = blockIdx.x;
  if (sel == nullptr || sel[row]) {
    float* xr = x + (size_t)row * DD;
    const float* yr = y + (size_t)row * DD;
    for (int i = threadIdx.x; i < DD; i += blockDim.x) xr[i] = yr[i];
  }
}

__global__ void topk_softmax_kernel(const float* __restrict__ sim,
                                    int* __restrict__ tidx,
                                    float* __restrict__ twts) {
  const int row = blockIdx.x;
  const int tid = threadIdx.x;
  const float* s = sim + (size_t)row * NMEM;
  __shared__ float bv[128];
  __shared__ int   bi[128];
  __shared__ int   chosen[TOPK];
  __shared__ float cval[TOPK];
  for (int it = 0; it < TOPK; ++it) {
    float best = -3.0e38f;
    int   bidx = 0x7fffffff;
    for (int k = tid; k < NMEM; k += blockDim.x) {
      bool used = false;
      for (int j = 0; j < it; ++j) used |= (chosen[j] == k);
      const float v = used ? -3.0e38f : s[k];
      if (v > best || (v == best && k < bidx)) { best = v; bidx = k; }
    }
    bv[tid] = best; bi[tid] = bidx; __syncthreads();
    for (int o = 64; o > 0; o >>= 1) {
      if (tid < o) {
        if (bv[tid + o] > bv[tid] ||
            (bv[tid + o] == bv[tid] && bi[tid + o] < bi[tid])) {
          bv[tid] = bv[tid + o]; bi[tid] = bi[tid + o];
        }
      }
      __syncthreads();
    }
    if (tid == 0) { chosen[it] = bi[0]; cval[it] = bv[0]; }
    __syncthreads();
  }
  if (tid == 0) {
    float mx = cval[0];
    for (int j = 1; j < TOPK; ++j) mx = fmaxf(mx, cval[j]);
    float e[TOPK], sum = 0.f;
    for (int j = 0; j < TOPK; ++j) { e[j] = __expf(cval[j] - mx); sum += e[j]; }
    const float inv = 1.f / sum;
    for (int j = 0; j < TOPK; ++j) {
      tidx[(size_t)row * TOPK + j] = chosen[j];
      twts[(size_t)row * TOPK + j] = e[j] * inv;
    }
  }
}

__global__ void gather_weighted_kernel(const float* __restrict__ memv,
                                       const int* __restrict__ tidx,
                                       const float* __restrict__ twts,
                                       float* __restrict__ out) {
  const int row = blockIdx.x;
  int   idx[TOPK];
  float wt[TOPK];
  for (int j = 0; j < TOPK; ++j) {
    idx[j] = tidx[(size_t)row * TOPK + j];
    wt[j]  = twts[(size_t)row * TOPK + j];
  }
  float* o = out + (size_t)row * DD;
  for (int i = threadIdx.x; i < DD; i += blockDim.x) {
    float acc = 0.f;
    for (int j = 0; j < TOPK; ++j) acc += wt[j] * memv[(size_t)idx[j] * DD + i];
    o[i] = acc;
  }
}

__global__ void concat_kernel(const float* __restrict__ a,
                              const float* __restrict__ b,
                              float* __restrict__ o) {
  const int row = blockIdx.x;
  float* orow = o + (size_t)row * 2 * DD;
  const float* ar = a + (size_t)row * DD;
  const float* br = b + (size_t)row * DD;
  for (int i = threadIdx.x; i < DD; i += blockDim.x) {
    orow[i]      = ar[i];
    orow[DD + i] = br[i];
  }
}

__global__ void bias_gelu_kernel(float* __restrict__ g,
                                 const float* __restrict__ bias,
                                 int N, long long n) {
  const long long i = (long long)blockIdx.x * blockDim.x + threadIdx.x;
  if (i < n) {
    const int c = (int)(i % N);
    const float v = g[i] + bias[c];
    g[i] = 0.5f * v * (1.f + erff(v * 0.70710678118654752f));
  }
}

__global__ void gate_dot_kernel(const float* __restrict__ g1,
                                const float* __restrict__ w2,
                                const float* __restrict__ b2,
                                float* __restrict__ gate, int N) {
  const int row = blockIdx.x;
  const int tid = threadIdx.x;
  const float* gr = g1 + (size_t)row * N;
  float s = 0.f;
  for (int i = tid; i < N; i += blockDim.x) s += gr[i] * w2[i];
  __shared__ float red[128];
  red[tid] = s; __syncthreads();
  for (int o = 64; o > 0; o >>= 1) {
    if (tid < o) red[tid] += red[tid + o];
    __syncthreads();
  }
  if (tid == 0) gate[row] = 1.f / (1.f + __expf(-(red[0] + b2[0])));
}

__global__ void gate_apply_kernel(float* __restrict__ x,
                                  const float* __restrict__ retr,
                                  const float* __restrict__ gate) {
  const int row = blockIdx.x;
  const float g = gate[row];
  float* xr = x + (size_t)row * DD;
  const float* rr = retr + (size_t)row * DD;
  for (int i = threadIdx.x; i < DD; i += blockDim.x) xr[i] += g * rr[i];
}

// ---------------------------------------------------------------------------
// Host orchestration
// ---------------------------------------------------------------------------
extern "C" void kernel_launch(void* const* d_in, const int* in_sizes, int n_in,
                              void* d_out, int out_size, void* d_ws, size_t ws_size,
                              hipStream_t stream) {
  (void)in_sizes; (void)n_in; (void)out_size; (void)ws_size;
  const int BT = BB * TT;

  const int*   input_ids    = (const int*)d_in[0];
  const float* embed_w      = (const float*)d_in[1];
  const float* pos_w        = (const float*)d_in[2];
  const float* qkv_w        = (const float*)d_in[3];
  const float* out_w        = (const float*)d_in[4];
  const float* norm1_w      = (const float*)d_in[5];
  const float* norm2_w      = (const float*)d_in[6];
  const float* ff_w1        = (const float*)d_in[7];
  const float* ff_w2        = (const float*)d_in[8];
  const float* ff_w3        = (const float*)d_in[9];
  const float* router_w     = (const float*)d_in[10];
  const float* lat_qkv_w    = (const float*)d_in[11];
  const float* lat_out_w    = (const float*)d_in[12];
  const float* lat_norm1_w  = (const float*)d_in[13];
  const float* lat_norm2_w  = (const float*)d_in[14];
  const float* lat_ff_w1    = (const float*)d_in[15];
  const float* lat_ff_w2    = (const float*)d_in[16];
  const float* lat_ff_w3    = (const float*)d_in[17];
  const float* mem_keys     = (const float*)d_in[18];
  const float* mem_values   = (const float*)d_in[19];
  const float* mem_qp       = (const float*)d_in[20];
  const float* mem_op       = (const float*)d_in[21];
  const float* gate_w1      = (const float*)d_in[22];
  const float* gate_b1      = (const float*)d_in[23];
  const float* gate_w2      = (const float*)d_in[24];
  const float* gate_b2      = (const float*)d_in[25];
  const float* final_norm_w = (const float*)d_in[26];

  // Workspace bump allocator (floats), with explicit aliasing where lifetimes
  // do not overlap.
  float* ws = (float*)d_ws;
  size_t off = 0;
  auto alloc = [&](size_t n) { float* p = ws + off; off += n; return p; };
  float* x    = alloc((size_t)BT * DD);
  float* hn   = alloc((size_t)BT * DD);
  float* qkv  = alloc((size_t)BT * 3 * DD);
  float* sc   = alloc((size_t)BB * HH * TT * TT);
  float* attn = alloc((size_t)BT * DD);
  float* xa   = alloc((size_t)BT * DD);
  float* h1   = alloc((size_t)BT * DFF);
  float* h2   = alloc((size_t)BT * DFF);
  float* y    = alloc((size_t)BT * DD);
  float* rsc  = alloc((size_t)BT);
  int*   sel  = (int*)alloc((size_t)BT);
  int*   tidx = (int*)alloc((size_t)BT * TOPK);
  float* twts = alloc((size_t)BT * TOPK);
  float* gate = alloc((size_t)BT);
  // Aliases (post-layer stage reuses layer scratch):
  float* qm    = hn;    // q = x @ mem_qp.T
  float* sim   = h1;    // [BT, NMEM]  (h1 is [BT, DFF] >= that)
  float* retr0 = attn;  // weighted memory values
  float* retr  = xa;    // projected retrieval
  float* gi    = qkv;   // [BT, 2D] concat (qkv is [BT, 3D])
  float* g1    = h2;    // [BT, D/2]

  const long long nBD  = (long long)BT * DD;
  const long long nBF  = (long long)BT * DFF;
  const int EB = 256;

  // x = embed[ids] + pos
  embed_kernel<<<BT, 256, 0, stream>>>(input_ids, embed_w, pos_w, x);

  auto run_tblock = [&](const float* qW, const float* oW, const float* n1W,
                        const float* n2W, const float* w1, const float* w2,
                        const float* w3, float* yout) {
    // hn = rmsnorm(x, n1)
    rmsnorm_kernel<<<BT, 256, 0, stream>>>(x, n1W, hn, DD);
    // qkv = hn @ qW.T   [BT, 3D]
    gemm_nt(stream, hn, qW, qkv, BT, 3 * DD, DD, DD, DD, 3 * DD,
            0, 0, 0, 0, 0, 0, 1, 1, 1.f);
    // scores[b,h] = Q Kt / sqrt(HD)
    gemm_nt(stream, qkv + 0, qkv + DD, sc, TT, TT, HD, 3 * DD, 3 * DD, TT,
            (long long)TT * 3 * DD, HD, (long long)TT * 3 * DD, HD,
            (long long)HH * TT * TT, (long long)TT * TT, BB * HH, HH, 0.125f);
    softmax_causal_kernel<<<BB * HH * TT, 256, 0, stream>>>(sc);
    // attn[b,t,h,:] = probs @ V
    gemm_nn(stream, sc, qkv + 2 * DD, attn, TT, HD, TT, TT, 3 * DD, DD,
            (long long)HH * TT * TT, (long long)TT * TT,
            (long long)TT * 3 * DD, HD, (long long)TT * DD, HD,
            BB * HH, HH, 1.f);
    // xa = x + attn @ oW.T
    gemm_nt(stream, attn, oW, xa, BT, DD, DD, DD, DD, DD,
            0, 0, 0, 0, 0, 0, 1, 1, 1.f);
    add_inplace_kernel<<<(int)((nBD + EB - 1) / EB), EB, 0, stream>>>(xa, x, nBD);
    // SwiGLU
    rmsnorm_kernel<<<BT, 256, 0, stream>>>(xa, n2W, hn, DD);
    gemm_nt(stream, hn, w1, h1, BT, DFF, DD, DD, DD, DFF,
            0, 0, 0, 0, 0, 0, 1, 1, 1.f);
    gemm_nt(stream, hn, w2, h2, BT, DFF, DD, DD, DD, DFF,
            0, 0, 0, 0, 0, 0, 1, 1, 1.f);
    silu_mul_kernel<<<(int)((nBF + EB - 1) / EB), EB, 0, stream>>>(h1, h2, nBF);
    gemm_nt(stream, h1, w3, yout, BT, DD, DFF, DFF, DFF, DD,
            0, 0, 0, 0, 0, 0, 1, 1, 1.f);
    add_inplace_kernel<<<(int)((nBD + EB - 1) / EB), EB, 0, stream>>>(yout, xa, nBD);
  };

  // ---- 12 transformer layers (odd layers are MoD) ----
  for (int l = 0; l < LL; ++l) {
    run_tblock(qkv_w + (size_t)l * 3 * DD * DD, out_w + (size_t)l * DD * DD,
               norm1_w + (size_t)l * DD, norm2_w + (size_t)l * DD,
               ff_w1 + (size_t)l * DFF * DD, ff_w2 + (size_t)l * DFF * DD,
               ff_w3 + (size_t)l * DD * DFF, y);
    if (l % 2 == 1) {  // MoD: router on pre-block x, keep y only at top-CAP
      rowdot_kernel<<<BT, 128, 0, stream>>>(x, router_w + (size_t)l * DD, rsc, DD);
      mod_select_kernel<<<BT, 128, 0, stream>>>(rsc, sel);
      mod_merge_kernel<<<BT, 256, 0, stream>>>(x, y, sel);
    } else {
      mod_merge_kernel<<<BT, 256, 0, stream>>>(x, y, (const int*)nullptr);
    }
  }

  // ---- latent reasoning: NLAT passes of the shared block ----
  for (int i = 0; i < NLAT; ++i) {
    run_tblock(lat_qkv_w, lat_out_w, lat_norm1_w, lat_norm2_w,
               lat_ff_w1, lat_ff_w2, lat_ff_w3, y);
    mod_merge_kernel<<<BT, 256, 0, stream>>>(x, y, (const int*)nullptr);
  }

  // ---- kNN memory retrieval ----
  gemm_nt(stream, x, mem_qp, qm, BT, DD, DD, DD, DD, DD,
          0, 0, 0, 0, 0, 0, 1, 1, 1.f);
  gemm_nt(stream, qm, mem_keys, sim, BT, NMEM, DD, DD, DD, NMEM,
          0, 0, 0, 0, 0, 0, 1, 1, 1.f / sqrtf((float)DD));
  topk_softmax_kernel<<<BT, 128, 0, stream>>>(sim, tidx, twts);
  gather_weighted_kernel<<<BT, 256, 0, stream>>>(mem_values, tidx, twts, retr0);
  gemm_nt(stream, retr0, mem_op, retr, BT, DD, DD, DD, DD, DD,
          0, 0, 0, 0, 0, 0, 1, 1, 1.f);

  // ---- gating ----
  concat_kernel<<<BT, 256, 0, stream>>>(x, retr, gi);
  gemm_nt(stream, gi, gate_w1, g1, BT, DD / 2, 2 * DD, 2 * DD, 2 * DD, DD / 2,
          0, 0, 0, 0, 0, 0, 1, 1, 1.f);
  const long long nG = (long long)BT * (DD / 2);
  bias_gelu_kernel<<<(int)((nG + EB - 1) / EB), EB, 0, stream>>>(g1, gate_b1,
                                                                DD / 2, nG);
  gate_dot_kernel<<<BT, 128, 0, stream>>>(g1, gate_w2, gate_b2, gate, DD / 2);
  gate_apply_kernel<<<BT, 256, 0, stream>>>(x, retr, gate);

  // ---- final norm + tied LM head ----
  rmsnorm_kernel<<<BT, 256, 0, stream>>>(x, final_norm_w, hn, DD);
  gemm_nt(stream, hn, embed_w, (float*)d_out, BT, VOC, DD, DD, DD, VOC,
          0, 0, 0, 0, 0, 0, 1, 1, 1.f);
}
